// MultiHeadAttention_71279277244764
// MI455X (gfx1250) — matrix operand order repaired and checked by reading, not compiled, not run on hardware
//
#include <hip/hip_runtime.h>
#include <hip/hip_bf16.h>
#include <cstdint>

typedef __bf16 bf16_t;
typedef __attribute__((ext_vector_type(16))) __bf16 v16bf;
typedef __attribute__((ext_vector_type(8)))  float  v8f;

namespace {
constexpr int kB = 2, kT = 2048, kD = 1024, kN = 16, kH = 64;
constexpr int kBT = kB * kT;   // 4096 rows of X
constexpr int kNH = kN * kH;   // 1024 qkv cols
constexpr float kLog2e = 1.442695041f;

// workspace layout (bytes)
constexpr size_t OFF_XB  = 0;                                 // X  bf16 [BT][D]
constexpr size_t OFF_WQT = OFF_XB  + (size_t)kBT * kD  * 2;   // WqT bf16 [NH][D]
constexpr size_t OFF_WKT = OFF_WQT + (size_t)kD  * kNH * 2;   // WkT bf16 [NH][D]
constexpr size_t OFF_WVT = OFF_WKT + (size_t)kD  * kNH * 2;   // WvT bf16 [NH][D]
constexpr size_t OFF_WOB = OFF_WVT + (size_t)kD  * kNH * 2;   // Wo  bf16 [D][NH]
constexpr size_t OFF_QB  = OFF_WOB + (size_t)kD  * kNH * 2;   // Q   bf16 [B][N][T][H]
constexpr size_t OFF_KB  = OFF_QB  + (size_t)kBT * kNH * 2;   // K   bf16 [B][N][T][H]
constexpr size_t OFF_VT  = OFF_KB  + (size_t)kBT * kNH * 2;   // Vt  bf16 [B][N][H][T]
constexpr size_t OFF_CB  = OFF_VT  + (size_t)kBT * kNH * 2;   // Ctx bf16 [B][T][N][H]

constexpr int kBK  = 32;        // GEMM K-slice per LDS stage
constexpr int kLDG = kBK + 8;   // padded GEMM LDS row stride (80B, 16B aligned)
constexpr int kLDK = 64 + 8;    // flash K-tile LDS stride
constexpr int kLDV = 32 + 8;    // flash V-tile LDS stride
} // namespace

// ---------- fragment helpers ----------

// Load 16 bf16 as two contiguous 16-byte chunks (A-frag / B-frag element order).
__device__ __forceinline__ v16bf load_frag32(const bf16_t* p0, const bf16_t* p1) {
  v16bf r;
  uint4* d = reinterpret_cast<uint4*>(&r);
  d[0] = *reinterpret_cast<const uint4*>(p0);
  d[1] = *reinterpret_cast<const uint4*>(p1);
  return r;
}

__device__ __forceinline__ v8f wmma_bf16(v16bf a, v16bf b, v8f c) {
  return __builtin_amdgcn_wmma_f32_16x16x32_bf16(
      /*neg_a=*/false, a, /*neg_b=*/false, b,
      /*c_mod=*/(short)0, c, /*reuse_a=*/false, /*reuse_b=*/false);
}

__device__ __forceinline__ v8f vzero8() {
  v8f z = {0.f, 0.f, 0.f, 0.f, 0.f, 0.f, 0.f, 0.f};
  return z;
}

// generic LDS pointer -> 32-bit DS offset (aperture lives in addr[63:32])
__device__ __forceinline__ unsigned lds_off32(const void* p) {
  return (unsigned)(unsigned long long)p;
}

// Stage a ROWS x COLS bf16 tile into LDS (row stride LSTRIDE) from
// gBase[(rowBase+row)*gRowStride + col0 + col] via the CDNA5 async-to-LDS DMA
// (ASYNCcnt). NT threads, 16B per async op, uniform trip counts (EXEC all-1s).
template <int ROWS, int COLS, int LSTRIDE, int NT>
__device__ __forceinline__ void async_fill(bf16_t* ldsTile, const bf16_t* gBase,
                                           size_t gRowStride, int rowBase, int col0,
                                           int tid) {
  constexpr int CHUNKS = ROWS * COLS / 8;   // 16B chunks
  constexpr int CPT    = CHUNKS / NT;       // chunks per thread
  constexpr int CPR    = COLS / 8;          // chunks per row
#pragma unroll
  for (int c = 0; c < CPT; ++c) {
    const int chunk = tid * CPT + c;
    const int row = chunk / CPR;
    const int col = (chunk % CPR) * 8;
    const unsigned lds  = lds_off32(ldsTile + row * LSTRIDE + col);
    const unsigned goff = (unsigned)(((size_t)(rowBase + row) * gRowStride + col0 + col) * 2);
    asm volatile("global_load_async_to_lds_b128 %0, %1, %2"
                 :: "v"(lds), "v"(goff), "s"(gBase)
                 : "memory");
  }
}

__device__ __forceinline__ void wait_async0() {
  asm volatile("s_wait_asynccnt 0" ::: "memory");
}

// ---------- pack kernels (fp32 -> bf16, optional transpose) ----------

__global__ void pack_convert(const float* __restrict__ src, bf16_t* __restrict__ dst, int n) {
  int i = blockIdx.x * blockDim.x + threadIdx.x;
  if (i < n) dst[i] = (bf16_t)src[i];
}

// src[rows][cols] -> dst[cols][rows] (coalesced on write)
__global__ void pack_transpose(const float* __restrict__ src, bf16_t* __restrict__ dst,
                               int rows, int cols) {
  int i = blockIdx.x * blockDim.x + threadIdx.x;
  if (i < rows * cols) {
    int r = i % rows;   // source row
    int c = i / rows;   // source col == dst row
    dst[i] = (bf16_t)src[r * cols + c];
  }
}

// ---------- LDS-staged, async double-buffered bf16 GEMM ----------
// C[4096 x 1024] = A[4096 x 1024] * B^T, B^T given row-major as Bg[1024 col][1024 k].
// Block tile 128(M) x 256(N), 8 waves, wave tile 64x64 (4x4 WMMA accumulators).
// MODE 0: Q epilogue (bias+softplus scale, [B][N][T][H])
// MODE 1: K epilogue (bias, [B][N][T][H])
// MODE 2: V epilogue (bias, transposed [B][N][H][T])
// MODE 3: output projection epilogue (bias, fp32 [BT][D])
template <int MODE>
__global__ __launch_bounds__(256) void gemm_bf16(
    const bf16_t* __restrict__ Ag,
    const bf16_t* __restrict__ Bg,
    const float*  __restrict__ bias,
    const float*  __restrict__ pds,
    bf16_t* __restrict__ outB,
    float*  __restrict__ outF) {
  __shared__ bf16_t sA[2][128 * kLDG];   // 20 KB
  __shared__ bf16_t sB[2][256 * kLDG];   // 40 KB

  const int tid   = threadIdx.x;
  const int lane  = tid & 31;
  const int wave  = tid >> 5;
  const int mBase = blockIdx.x * 128;
  const int nBase = blockIdx.y * 256;
  const int waveM = (wave >> 2) * 64;    // 2 wave rows
  const int waveN = (wave & 3) * 64;     // 4 wave cols
  const int l16  = lane & 15;
  const int hi8  = (lane >> 4) * 8;
  const int hi16 = (lane >> 4) * 16;

  v8f acc[4][4];
#pragma unroll
  for (int mt = 0; mt < 4; ++mt)
#pragma unroll
    for (int nf = 0; nf < 4; ++nf) acc[mt][nf] = vzero8();

  // prologue: stage first K-slice
  async_fill<128, kBK, kLDG, 256>(sA[0], Ag, kD, mBase, 0, tid);
  async_fill<256, kBK, kLDG, 256>(sB[0], Bg, kD, nBase, 0, tid);
  wait_async0();
  __syncthreads();

  constexpr int nK = kD / kBK;  // 32
  for (int kb = 0; kb < nK; ++kb) {
    const int cur = kb & 1;
    if (kb + 1 < nK) {  // DMA next slice into the other buffer while computing
      async_fill<128, kBK, kLDG, 256>(sA[cur ^ 1], Ag, kD, mBase, (kb + 1) * kBK, tid);
      async_fill<256, kBK, kLDG, 256>(sB[cur ^ 1], Bg, kD, nBase, (kb + 1) * kBK, tid);
    }

    const bf16_t* a0 = sA[cur];
    const bf16_t* b0 = sB[cur];
    v16bf af[4];
#pragma unroll
    for (int mt = 0; mt < 4; ++mt) {
      const bf16_t* p = a0 + (waveM + mt * 16 + l16) * kLDG;
      af[mt] = load_frag32(p + hi8, p + 16 + hi8);
    }
#pragma unroll
    for (int nf = 0; nf < 4; ++nf) {
      const bf16_t* p = b0 + (waveN + nf * 16 + l16) * kLDG + hi8;
      const v16bf bfg = load_frag32(p, p + 16);
#pragma unroll
      for (int mt = 0; mt < 4; ++mt)
        acc[mt][nf] = wmma_bf16(af[mt], bfg, acc[mt][nf]);
    }

    wait_async0();     // next slice resident (this wave's DMA)
    __syncthreads();   // everyone done reading cur + everyone's DMA done
  }

  // epilogue: C/D layout rows M = r + hi8, cols N = l16
#pragma unroll
  for (int mt = 0; mt < 4; ++mt) {
#pragma unroll
    for (int nf = 0; nf < 4; ++nf) {
      const int col = nBase + waveN + nf * 16 + l16;
      const float bs = bias[col];
      float sc = 1.0f;
      if (MODE == 0) sc = (kLog2e * 0.125f) * log1pf(expf(pds[col % kH]));
#pragma unroll
      for (int r = 0; r < 8; ++r) {
        const int mrow = mBase + waveM + mt * 16 + r + hi8;
        float v = acc[mt][nf][r] + bs;
        if (MODE == 0) v *= sc;
        if (MODE == 3) {
          outF[(size_t)mrow * kD + col] = v;
        } else {
          const int b = mrow / kT, t = mrow % kT;
          const int n = col / kH, h = col % kH;
          if (MODE == 2)
            outB[(((size_t)b * kN + n) * kH + h) * kT + t] = (bf16_t)v;  // V transposed
          else
            outB[(((size_t)b * kN + n) * kT + t) * kH + h] = (bf16_t)v;
        }
      }
    }
  }
}

// ---------- flash attention ----------
// Block = 4 waves sharing one (b,n); each wave owns 16 query rows. K/V chunks of
// 32 keys are double-buffered in LDS via async DMA and shared by all 4 waves.
__global__ __launch_bounds__(128) void flash_attn(
    const bf16_t* __restrict__ Qb,   // [B][N][T][H]
    const bf16_t* __restrict__ Kb,   // [B][N][T][H]
    const bf16_t* __restrict__ Vt,   // [B][N][H][T]
    const int*    __restrict__ mask, // [B][1][T][T]
    bf16_t* __restrict__ Cb) {       // [B][T][N][H]
  __shared__ bf16_t sK[2][32 * kLDK];   // 32 keys x 64 h   (9 KB)
  __shared__ bf16_t sV[2][64 * kLDV];   // 64 h x 32 keys   (10 KB)
  __shared__ bf16_t plds[4][16][32];    // per-wave P tile  (4 KB)

  const int tid   = threadIdx.x;
  const int lane  = tid & 31;
  const int wave  = tid >> 5;
  const int tile  = blockIdx.x * 4 + wave;   // 4096 tiles
  const int qTile = tile & 127;              // T/16 = 128
  const int bn    = tile >> 7;               // 32 (b,n) pairs (same for all 4 waves)
  const int n = bn & 15, b = bn >> 4;
  const int tBase = qTile * 16;
  const int l16  = lane & 15;
  const int hi8  = (lane >> 4) * 8;
  const int hi16 = (lane >> 4) * 16;

  // Q A-fragments covering K-dim H=64 (two K=32 chunks), register resident
  const bf16_t* qRow = Qb + (((size_t)b * kN + n) * kT + (tBase + l16)) * kH;
  v16bf qf0 = load_frag32(qRow + hi8,      qRow + 16 + hi8);
  v16bf qf1 = load_frag32(qRow + 32 + hi8, qRow + 48 + hi8);

  const bf16_t* Kbn  = Kb + ((size_t)b * kN + n) * kT * kH;
  const bf16_t* Vbn  = Vt + ((size_t)b * kN + n) * kH * kT;
  const int*    mrow = mask + (size_t)b * kT * kT;

  float m[8], l[8];
  v8f accO[4];
#pragma unroll
  for (int r = 0; r < 8; ++r) { m[r] = -1.0e30f; l[r] = 0.f; }
  accO[0] = vzero8(); accO[1] = vzero8(); accO[2] = vzero8(); accO[3] = vzero8();

  // prologue: stage first key chunk
  async_fill<32, 64, kLDK, 128>(sK[0], Kbn, kH, 0, 0, tid);
  async_fill<64, 32, kLDV, 128>(sV[0], Vbn, kT, 0, 0, tid);
  wait_async0();
  __syncthreads();

  constexpr int nS = kT / 32;  // 64 key chunks
  for (int scnk = 0; scnk < nS; ++scnk) {
    const int s0  = scnk * 32;
    const int cur = scnk & 1;
    if (scnk + 1 < nS) {  // DMA next K/V chunk while computing this one
      async_fill<32, 64, kLDK, 128>(sK[cur ^ 1], Kbn, kH, s0 + 32, 0, tid);
      async_fill<64, 32, kLDV, 128>(sV[cur ^ 1], Vbn, kT, 0, s0 + 32, tid);
    }
    const bf16_t* sKc = sK[cur];
    const bf16_t* sVc = sV[cur];

    // scores for two 16x16 tiles (keys s0..+15, s0+16..+31), operands from LDS
    v8f sc0 = vzero8(), sc1 = vzero8();
    {
      const bf16_t* kp0 = sKc + l16 * kLDK + hi8;
      const bf16_t* kp1 = sKc + (16 + l16) * kLDK + hi8;
      sc0 = wmma_bf16(qf0, load_frag32(kp0, kp0 + 16), sc0);
      sc1 = wmma_bf16(qf0, load_frag32(kp1, kp1 + 16), sc1);
      sc0 = wmma_bf16(qf1, load_frag32(kp0 + 32, kp0 + 48), sc0);
      sc1 = wmma_bf16(qf1, load_frag32(kp1 + 32, kp1 + 48), sc1);
    }

    // mask + online softmax (per-row stats replicated across 16-lane halves)
#pragma unroll
    for (int r = 0; r < 8; ++r) {
      const int t = tBase + r + hi8;
      const int* mr = mrow + (size_t)t * kT + s0;
      float x0 = (mr[l16] == 0)      ? -1.0e9f : sc0[r];
      float x1 = (mr[16 + l16] == 0) ? -1.0e9f : sc1[r];
      float mx = fmaxf(x0, x1);
#pragma unroll
      for (int off = 1; off < 16; off <<= 1) mx = fmaxf(mx, __shfl_xor(mx, off, 16));
      const float mNew = fmaxf(m[r], mx);
      const float ef = __expf(m[r] - mNew);
      const float p0 = __expf(x0 - mNew);
      const float p1 = __expf(x1 - mNew);
      float ps = p0 + p1;
#pragma unroll
      for (int off = 1; off < 16; off <<= 1) ps += __shfl_xor(ps, off, 16);
      l[r] = l[r] * ef + ps;
      m[r] = mNew;
      accO[0][r] *= ef; accO[1][r] *= ef; accO[2][r] *= ef; accO[3][r] *= ef;
      // P tile: C/D layout -> LDS (row M, col = key offset)
      plds[wave][r + hi8][l16]      = (bf16_t)p0;
      plds[wave][r + hi8][16 + l16] = (bf16_t)p1;
    }

    // re-load P in A-fragment layout (intra-wave; DScnt ordering handled by compiler)
    v16bf pf = load_frag32(&plds[wave][l16][hi8], &plds[wave][l16][16 + hi8]);

    // O += P(16x32) x V(32x64): 4 WMMAs over H-tiles, V^T tile in LDS
#pragma unroll
    for (int j = 0; j < 4; ++j) {
      const bf16_t* vp = sVc + (j * 16 + l16) * kLDV + hi8;
      accO[j] = wmma_bf16(pf, load_frag32(vp, vp + 16), accO[j]);
    }

    wait_async0();     // next chunk resident (this wave's DMA)
    __syncthreads();   // everyone done reading cur
  }

  // normalize and store context as [B][T][N][H] bf16 (A layout for output GEMM)
#pragma unroll
  for (int j = 0; j < 4; ++j) {
#pragma unroll
    for (int r = 0; r < 8; ++r) {
      const int t = tBase + r + hi8;
      const float v = accO[j][r] / l[r];
      Cb[(((size_t)b * kT + t) * kN + n) * kH + j * 16 + l16] = (bf16_t)v;
    }
  }
}

// ---------- host ----------
extern "C" void kernel_launch(void* const* d_in, const int* in_sizes, int n_in,
                              void* d_out, int out_size, void* d_ws, size_t ws_size,
                              hipStream_t stream) {
  const float* x    = (const float*)d_in[0];
  const int*   mask = (const int*)d_in[1];
  const float* wq   = (const float*)d_in[2];
  const float* bq   = (const float*)d_in[3];
  const float* wk   = (const float*)d_in[4];
  const float* bk   = (const float*)d_in[5];
  const float* wv   = (const float*)d_in[6];
  const float* bv   = (const float*)d_in[7];
  const float* wo   = (const float*)d_in[8];
  const float* bo   = (const float*)d_in[9];
  const float* pds  = (const float*)d_in[10];

  char* ws = (char*)d_ws;
  bf16_t* Xb  = (bf16_t*)(ws + OFF_XB);
  bf16_t* WqT = (bf16_t*)(ws + OFF_WQT);
  bf16_t* WkT = (bf16_t*)(ws + OFF_WKT);
  bf16_t* WvT = (bf16_t*)(ws + OFF_WVT);
  bf16_t* Wob = (bf16_t*)(ws + OFF_WOB);
  bf16_t* Qb  = (bf16_t*)(ws + OFF_QB);
  bf16_t* Kbb = (bf16_t*)(ws + OFF_KB);
  bf16_t* Vtb = (bf16_t*)(ws + OFF_VT);
  bf16_t* Cb  = (bf16_t*)(ws + OFF_CB);

  // pack to bf16 (X straight, Wq/Wk/Wv transposed, Wo straight)
  pack_convert<<<(kBT * kD + 255) / 256, 256, 0, stream>>>(x, Xb, kBT * kD);
  pack_transpose<<<(kD * kNH + 255) / 256, 256, 0, stream>>>(wq, WqT, kD, kNH);
  pack_transpose<<<(kD * kNH + 255) / 256, 256, 0, stream>>>(wk, WkT, kD, kNH);
  pack_transpose<<<(kD * kNH + 255) / 256, 256, 0, stream>>>(wv, WvT, kD, kNH);
  pack_convert<<<(kD * kNH + 255) / 256, 256, 0, stream>>>(wo, Wob, kD * kNH);

  dim3 ggrid(kBT / 128, kNH / 256);  // 32 x 4
  // QKV projections (async-LDS staged WMMA)
  gemm_bf16<0><<<ggrid, 256, 0, stream>>>(Xb, WqT, bq, pds, Qb, nullptr);
  gemm_bf16<1><<<ggrid, 256, 0, stream>>>(Xb, WkT, bk, pds, Kbb, nullptr);
  gemm_bf16<2><<<ggrid, 256, 0, stream>>>(Xb, WvT, bv, pds, Vtb, nullptr);

  // streaming-softmax attention (WMMA QK^T and PV, async-LDS staged K/V)
  flash_attn<<<1024, 128, 0, stream>>>(Qb, Kbb, Vtb, mask, Cb);

  // output projection (async-LDS staged WMMA)
  gemm_bf16<3><<<ggrid, 256, 0, stream>>>(Cb, Wob, bo, nullptr, nullptr, (float*)d_out);
}
